// Attention_3539053052668
// MI455X (gfx1250) — matrix operand order repaired and checked by reading, not compiled, not run on hardware
//
#include <hip/hip_runtime.h>
#include <math.h>

// ---------------------------------------------------------------------------
// Problem constants (match reference)
// ---------------------------------------------------------------------------
constexpr int Bb = 2, Nn = 2048, Ee = 1024, Hh = 16, Dd = 64;
constexpr int Mrows = Bb * Nn;          // 4096
constexpr int QKVC  = 3 * Hh * Dd;      // 3072

typedef _Float16 h16;
typedef __attribute__((ext_vector_type(16))) _Float16 v16h;
typedef __attribute__((ext_vector_type(8)))  _Float16 v8h;
typedef __attribute__((ext_vector_type(4)))  _Float16 v4h;
typedef __attribute__((ext_vector_type(8)))  float    v8f;

#define WMMA_F16(a, b, c) \
  __builtin_amdgcn_wmma_f32_16x16x32_f16(false, (a), false, (b), (short)0, (c), false, false)

static __device__ __forceinline__ v16h cat8(v8h lo, v8h hi) {
  return __builtin_shufflevector(lo, hi, 0, 1, 2, 3, 4, 5, 6, 7,
                                         8, 9, 10, 11, 12, 13, 14, 15);
}

// ===========================================================================
// Kernel 1: qkv = x @ Wqkv^T + bqkv  -> split into q (scaled), k, v  [B,H,N,D] f16
//   M = 4096 (B*N), N = 3072 (3*H*D), K = 1024 (E)
//   128x128 block tile, 8 waves; each wave 64x32 -> 4x2 WMMA tiles.
// ===========================================================================
__global__ __launch_bounds__(256) void qkv_proj_kernel(
    const float* __restrict__ x, const float* __restrict__ Wqkv,
    const float* __restrict__ bqkv,
    h16* __restrict__ qo, h16* __restrict__ ko, h16* __restrict__ vo) {
  __shared__ h16 sA[128][40];   // rows of x tile, k contiguous (pad -> 80B stride)
  __shared__ h16 sB[128][40];   // rows of Wqkv tile (= output cols), k contiguous

  const int t = threadIdx.x;
  const int lane = t & 31, w = t >> 5;
  const int wm = w >> 2, wn = w & 3;      // 2x4 wave grid
  const int hi = lane >> 4, ln = lane & 15;
  const int m0 = blockIdx.x * 128, n0 = blockIdx.y * 128;

  v8f acc[4][2];
#pragma unroll
  for (int mt = 0; mt < 4; ++mt)
#pragma unroll
    for (int nt = 0; nt < 2; ++nt) acc[mt][nt] = {};

  const int lr = t >> 3;          // 0..31
  const int lc = (t & 7) * 4;     // 0..28 step 4

  for (int k0 = 0; k0 < Ee; k0 += 32) {
#pragma unroll
    for (int p = 0; p < 4; ++p) {
      const int row = p * 32 + lr;
      float4 f = *(const float4*)(x + (size_t)(m0 + row) * Ee + k0 + lc);
      v4h hv = {(h16)f.x, (h16)f.y, (h16)f.z, (h16)f.w};
      *(v4h*)&sA[row][lc] = hv;
      float4 g = *(const float4*)(Wqkv + (size_t)(n0 + row) * Ee + k0 + lc);
      v4h gv = {(h16)g.x, (h16)g.y, (h16)g.z, (h16)g.w};
      *(v4h*)&sB[row][lc] = gv;
    }
    __syncthreads();

    v16h bfr[2];
#pragma unroll
    for (int nt = 0; nt < 2; ++nt) {
      const h16* bp = &sB[wn * 32 + nt * 16 + ln][hi * 16];
      bfr[nt] = cat8(*(const v8h*)bp, *(const v8h*)(bp + 8));
    }
#pragma unroll
    for (int mt = 0; mt < 4; ++mt) {
      const h16* ap = &sA[wm * 64 + mt * 16 + ln][0];
      v16h afr = cat8(*(const v8h*)(ap + hi * 16), *(const v8h*)(ap + hi * 16 + 8));
#pragma unroll
      for (int nt = 0; nt < 2; ++nt)
        acc[mt][nt] = WMMA_F16(afr, bfr[nt], acc[mt][nt]);
    }
    __syncthreads();
  }

  // Epilogue: column c -> (h, d, s) with c = h*3*D + d*3 + s   (reshape (H,D,3))
#pragma unroll
  for (int mt = 0; mt < 4; ++mt)
#pragma unroll
    for (int nt = 0; nt < 2; ++nt)
#pragma unroll
      for (int r = 0; r < 8; ++r) {
        const int mg = m0 + wm * 64 + mt * 16 + r + hi * 8;
        const int cg = n0 + wn * 32 + nt * 16 + ln;
        float val = acc[mt][nt][r] + bqkv[cg];
        const int s = cg % 3;
        const int d = (cg / 3) % Dd;
        const int hh = cg / (3 * Dd);
        const int b = mg >> 11;       // /2048
        const int n = mg & 2047;
        const size_t idx = (((size_t)b * Hh + hh) * Nn + n) * Dd + d;
        if (s == 0)      qo[idx] = (h16)(val * 0.125f);   // fold 1/sqrt(64)
        else if (s == 1) ko[idx] = (h16)val;
        else             vo[idx] = (h16)val;
      }
}

// ===========================================================================
// Kernel 2: causal flash attention.
//   grid = (N/128, B*H), block = 256 (8 waves). Wave w owns 16 q rows.
//   Key/value streamed in 32-row blocks through LDS. Online softmax.
// ===========================================================================
__global__ __launch_bounds__(256) void flash_attn_kernel(
    const h16* __restrict__ qin, const h16* __restrict__ kin,
    const h16* __restrict__ vin, h16* __restrict__ O) {
  __shared__ h16 sK[32][72];        // [j][d]  (d contiguous; 144B stride)
  __shared__ h16 sV[64][40];        // [d][j]  (j contiguous; 80B stride)
  __shared__ h16 sP[8][16][40];     // per-wave P bounce buffer

  const int t = threadIdx.x, lane = t & 31, w = t >> 5;
  const int hi = lane >> 4, ln = lane & 15;
  const int qb = blockIdx.x;
  const int bh = blockIdx.y;                         // b*H + h
  const size_t base = (size_t)bh * Nn * Dd;
  const int r0g = qb * 128 + w * 16;                 // wave's first q row

  // Q fragments (A-layout), reused across all key blocks. Scale pre-folded.
  const h16* qrow = qin + base + (size_t)(r0g + ln) * Dd;
  v16h aQ[2];
#pragma unroll
  for (int kk = 0; kk < 2; ++kk)
    aQ[kk] = cat8(*(const v8h*)(qrow + kk * 32 + hi * 16),
                  *(const v8h*)(qrow + kk * 32 + hi * 16 + 8));

  v8f accO[4];
#pragma unroll
  for (int dt = 0; dt < 4; ++dt) accO[dt] = {};
  float mrow[8], lrow[8];
#pragma unroll
  for (int r = 0; r < 8; ++r) { mrow[r] = -__builtin_inff(); lrow[r] = 0.f; }

  const int nblk = (qb + 1) * 4;        // causal: only blocks j0 <= qb*128+127
  const int ljr = t >> 3;               // 0..31 : key row within block
  const int ljc = (t & 7) * 8;          // 0..56 : d offset

  for (int bi = 0; bi < nblk; ++bi) {
    const int j0 = bi * 32;
    // Stage K tile row-major and V tile transposed ([d][j]) into LDS.
    {
      const h16* kp = kin + base + (size_t)(j0 + ljr) * Dd + ljc;
      *(v8h*)&sK[ljr][ljc] = *(const v8h*)kp;
      const h16* vp = vin + base + (size_t)(j0 + ljr) * Dd + ljc;
      v8h vv = *(const v8h*)vp;
#pragma unroll
      for (int e = 0; e < 8; ++e) sV[ljc + e][ljr] = vv[e];
    }
    __syncthreads();

    if (j0 <= r0g + 15) {               // wave-uniform: skip fully-masked blocks
      v8f accS[2] = {{}, {}};
#pragma unroll
      for (int nt = 0; nt < 2; ++nt)
#pragma unroll
        for (int kk = 0; kk < 2; ++kk) {
          const h16* bp = &sK[nt * 16 + ln][kk * 32 + hi * 16];
          v16h bK = cat8(*(const v8h*)bp, *(const v8h*)(bp + 8));
          accS[nt] = WMMA_F16(aQ[kk], bK, accS[nt]);
        }
      // Causal mask (only bites on the diagonal block)
#pragma unroll
      for (int nt = 0; nt < 2; ++nt)
#pragma unroll
        for (int r = 0; r < 8; ++r) {
          const int rg = r0g + r + hi * 8;
          const int cg = j0 + nt * 16 + ln;
          if (cg > rg) accS[nt][r] = -__builtin_inff();
        }
      // Online softmax (row reductions across the 16-lane group via shfl_xor)
      float alpha[8];
#pragma unroll
      for (int r = 0; r < 8; ++r) {
        float s0 = accS[0][r], s1 = accS[1][r];
        float mx = fmaxf(s0, s1);
        mx = fmaxf(mx, __shfl_xor(mx, 1));
        mx = fmaxf(mx, __shfl_xor(mx, 2));
        mx = fmaxf(mx, __shfl_xor(mx, 4));
        mx = fmaxf(mx, __shfl_xor(mx, 8));
        const float mn = fmaxf(mrow[r], mx);
        alpha[r] = __expf(mrow[r] - mn);
        const float p0 = __expf(s0 - mn), p1 = __expf(s1 - mn);
        float rs = p0 + p1;
        rs += __shfl_xor(rs, 1);
        rs += __shfl_xor(rs, 2);
        rs += __shfl_xor(rs, 4);
        rs += __shfl_xor(rs, 8);
        lrow[r] = lrow[r] * alpha[r] + rs;
        mrow[r] = mn;
        const int rl = r + hi * 8;
        sP[w][rl][ln]      = (h16)p0;   // C-layout -> LDS (per-wave, no barrier)
        sP[w][rl][16 + ln] = (h16)p1;
      }
#pragma unroll
      for (int dt = 0; dt < 4; ++dt)
#pragma unroll
        for (int r = 0; r < 8; ++r) accO[dt][r] *= alpha[r];

      // P (A-layout from LDS) x V
      const h16* pp = &sP[w][ln][0];
      v16h aP = cat8(*(const v8h*)(pp + hi * 16), *(const v8h*)(pp + hi * 16 + 8));
#pragma unroll
      for (int dt = 0; dt < 4; ++dt) {
        const h16* bp = &sV[dt * 16 + ln][hi * 16];
        v16h bV = cat8(*(const v8h*)bp, *(const v8h*)(bp + 8));
        accO[dt] = WMMA_F16(aP, bV, accO[dt]);
      }
    }
    __syncthreads();
  }

  // Epilogue: O[b, n, h*D + d] f16 (row-major [4096, 1024] for the final GEMM)
  const int b = bh / Hh, hh = bh % Hh;
#pragma unroll
  for (int dt = 0; dt < 4; ++dt)
#pragma unroll
    for (int r = 0; r < 8; ++r) {
      const int rg = r0g + r + hi * 8;
      const float ov = accO[dt][r] / lrow[r];
      O[((size_t)(b * Nn + rg)) * (Hh * Dd) + hh * Dd + dt * 16 + ln] = (h16)ov;
    }
}

// ===========================================================================
// Kernel 3: out = O @ Wout^T + bout   (M=4096, N=1024, K=1024), fp32 out.
// ===========================================================================
__global__ __launch_bounds__(256) void out_proj_kernel(
    const h16* __restrict__ Oin, const float* __restrict__ Wout,
    const float* __restrict__ bout, float* __restrict__ out) {
  __shared__ h16 sA[128][40];
  __shared__ h16 sB[128][40];

  const int t = threadIdx.x;
  const int lane = t & 31, w = t >> 5;
  const int wm = w >> 2, wn = w & 3;
  const int hi = lane >> 4, ln = lane & 15;
  const int m0 = blockIdx.x * 128, n0 = blockIdx.y * 128;
  const int K = Hh * Dd;     // 1024

  v8f acc[4][2];
#pragma unroll
  for (int mt = 0; mt < 4; ++mt)
#pragma unroll
    for (int nt = 0; nt < 2; ++nt) acc[mt][nt] = {};

  const int lrA = t >> 2;          // 0..63
  const int lcA = (t & 3) * 8;     // 0..24 step 8
  const int lrB = t >> 3;          // 0..31
  const int lcB = (t & 7) * 4;     // 0..28 step 4

  for (int k0 = 0; k0 < K; k0 += 32) {
#pragma unroll
    for (int p = 0; p < 2; ++p) {
      const int row = p * 64 + lrA;
      *(v8h*)&sA[row][lcA] =
          *(const v8h*)(Oin + (size_t)(m0 + row) * K + k0 + lcA);
    }
#pragma unroll
    for (int p = 0; p < 4; ++p) {
      const int row = p * 32 + lrB;
      float4 g = *(const float4*)(Wout + (size_t)(n0 + row) * K + k0 + lcB);
      v4h gv = {(h16)g.x, (h16)g.y, (h16)g.z, (h16)g.w};
      *(v4h*)&sB[row][lcB] = gv;
    }
    __syncthreads();

    v16h bfr[2];
#pragma unroll
    for (int nt = 0; nt < 2; ++nt) {
      const h16* bp = &sB[wn * 32 + nt * 16 + ln][hi * 16];
      bfr[nt] = cat8(*(const v8h*)bp, *(const v8h*)(bp + 8));
    }
#pragma unroll
    for (int mt = 0; mt < 4; ++mt) {
      const h16* ap = &sA[wm * 64 + mt * 16 + ln][0];
      v16h afr = cat8(*(const v8h*)(ap + hi * 16), *(const v8h*)(ap + hi * 16 + 8));
#pragma unroll
      for (int nt = 0; nt < 2; ++nt)
        acc[mt][nt] = WMMA_F16(afr, bfr[nt], acc[mt][nt]);
    }
    __syncthreads();
  }

#pragma unroll
  for (int mt = 0; mt < 4; ++mt)
#pragma unroll
    for (int nt = 0; nt < 2; ++nt)
#pragma unroll
      for (int r = 0; r < 8; ++r) {
        const int mg = m0 + wm * 64 + mt * 16 + r + hi * 8;
        const int cg = n0 + wn * 32 + nt * 16 + ln;
        out[(size_t)mg * Ee + cg] = acc[mt][nt][r] + bout[cg];
      }
}

// ===========================================================================
extern "C" void kernel_launch(void* const* d_in, const int* in_sizes, int n_in,
                              void* d_out, int out_size, void* d_ws, size_t ws_size,
                              hipStream_t stream) {
  const float* x    = (const float*)d_in[0];
  const float* Wqkv = (const float*)d_in[1];
  const float* bqkv = (const float*)d_in[2];
  const float* Wout = (const float*)d_in[3];
  const float* bout = (const float*)d_in[4];
  float* out = (float*)d_out;

  // Workspace layout (f16): q, k, v each [B,H,N,D], then O [B*N, H*D]
  const size_t QKVSZ = (size_t)Bb * Hh * Nn * Dd;   // 4,194,304 elements
  h16* q = (h16*)d_ws;
  h16* k = q + QKVSZ;
  h16* v = k + QKVSZ;
  h16* O = v + QKVSZ;

  qkv_proj_kernel<<<dim3(Mrows / 128, QKVC / 128), 256, 0, stream>>>(
      x, Wqkv, bqkv, q, k, v);
  flash_attn_kernel<<<dim3(Nn / 128, Bb * Hh), 256, 0, stream>>>(q, k, v, O);
  out_proj_kernel<<<dim3(Mrows / 128, Ee / 128), 256, 0, stream>>>(
      O, Wout, bout, out);
}
